// TopoEncoder_36318243455409
// MI455X (gfx1250) — matrix operand order repaired and checked by reading, not compiled, not run on hardware
//
#include <hip/hip_runtime.h>
#include <hip/hip_bf16.h>

// ---------------------------------------------------------------------------
// TopoEncoder forward on MI455X (gfx1250, wave32, WMMA 16x16x32 f16->f32)
// ---------------------------------------------------------------------------

#define B_N   65536
#define DIN_N 32
#define HID_N 128
#define LAT_N 32
#define NC_N  8
#define KC_N  512
#define MID_N 16

typedef __attribute__((ext_vector_type(16))) _Float16 v16h;
typedef __attribute__((ext_vector_type(8)))  float    v8f;
typedef __attribute__((__vector_size__(4 * sizeof(int)))) int i4v;

union V16H { v16h v; _Float16 h[16]; uint4 q[2]; };

#if defined(__has_builtin)
#if __has_builtin(__builtin_amdgcn_global_load_async_to_lds_b128) && \
    __has_builtin(__builtin_amdgcn_s_wait_asynccnt)
#define HAVE_ASYNC_LDS 1
#endif
#endif

__device__ inline v8f wmma_f16(v16h a, v16h b, v8f c) {
  return __builtin_amdgcn_wmma_f32_16x16x32_f16(false, a, false, b, (short)0, c,
                                                false, false);
}

__device__ inline float gelu_f(float x) {
  return 0.5f * x * (1.0f + erff(x * 0.70710678118654752440f));
}

// A-matrix (16x32, f16, K=32 tile) from row-major f16 buffer (LDS or global).
// Lane l<16: row l, K {0..15}; lane>=16: row l-16, K {16..31}.
__device__ inline v16h load_A_h16(const _Float16* rowsBase, int rowStrideH, int lane) {
  int row = lane & 15;
  int kb  = (lane < 16) ? 0 : 16;
  const _Float16* p = rowsBase + (size_t)row * rowStrideH + kb;
  V16H u;
  u.q[0] = *(const uint4*)(p);
  u.q[1] = *(const uint4*)(p + 8);
  return u.v;
}

// Same A layout, converting from row-major f32 source on the fly.
__device__ inline v16h load_A_f32(const float* rowsBase, int rowStrideF, int lane) {
  int row = lane & 15;
  int kb  = (lane < 16) ? 0 : 16;
  const float* p = rowsBase + (size_t)row * rowStrideF + kb;
  V16H u;
#pragma unroll
  for (int i = 0; i < 8; ++i) { u.h[i] = (_Float16)p[i]; u.h[8 + i] = (_Float16)p[8 + i]; }
  return u.v;
}

// B-matrix (32x16, f16, K=32 tile) from transposed weights wT[n][k] (ldk halfs).
__device__ inline v16h load_B_h16(const _Float16* wT, int ldk, int n0, int k0, int lane) {
  const _Float16* p = wT + (size_t)(n0 + (lane & 15)) * ldk + k0 + ((lane < 16) ? 0 : 16);
  V16H u;
  u.q[0] = *(const uint4*)(p);
  u.q[1] = *(const uint4*)(p + 8);
  return u.v;
}

// B-matrix from global f32 with contiguous K (w laid out [n][k], ldk floats).
__device__ inline v16h load_B_f32c(const float* w, int ldk, int n0, int k0, int lane) {
  const float* p = w + (size_t)(n0 + (lane & 15)) * ldk + k0 + ((lane < 16) ? 0 : 16);
  V16H u;
#pragma unroll
  for (int i = 0; i < 16; ++i) u.h[i] = (_Float16)p[i];
  return u.v;
}

// B-matrix from global f32 strided K (w laid out [k][n], row stride ldn floats).
__device__ inline v16h load_B_f32s(const float* w, int ldn, int n0, int k0, int lane) {
  const float* p = w + (size_t)(k0 + ((lane < 16) ? 0 : 16)) * ldn + n0 + (lane & 15);
  V16H u;
#pragma unroll
  for (int i = 0; i < 16; ++i) u.h[i] = (_Float16)p[(size_t)i * ldn];
  return u.v;
}

// ---------------------------------------------------------------------------
// K0: one-time codebook f32 -> f16 conversion + per-code norms
// ---------------------------------------------------------------------------
__global__ __launch_bounds__(256) void k0_prep(const float* __restrict__ codebook,
                                               _Float16* __restrict__ cb_h,
                                               float* __restrict__ cbn_g)
{
  int code = blockIdx.x * 256 + threadIdx.x;  // NC*KC = 4096 codes
  if (code < NC_N * KC_N) {
    const float* src = codebook + (size_t)code * LAT_N;
    _Float16* dst = cb_h + (size_t)code * LAT_N;
    float s = 0.f;
#pragma unroll
    for (int l = 0; l < LAT_N; ++l) {
      _Float16 h = (_Float16)src[l];
      dst[l] = h;
      float f = (float)h;
      s += f * f;
    }
    cbn_g[code] = s;
  }
}

// ---------------------------------------------------------------------------
// K1: encoder  x -> features -> v -> softmax charts -> c_bar, v_local
//     2 waves x 2 subtiles x 16 rows = 64 rows/block
// ---------------------------------------------------------------------------
__global__ __launch_bounds__(64) void k1_encoder(
    const float* __restrict__ x,
    const float* __restrict__ fe_w1, const float* __restrict__ fe_b1,
    const float* __restrict__ fe_w2, const float* __restrict__ fe_b2,
    const float* __restrict__ vp_w,  const float* __restrict__ vp_b,
    const float* __restrict__ cc,
    float* __restrict__ enc_out, int* __restrict__ kch_out,
    float* __restrict__ cbar_out,
    float* __restrict__ vloc_f32, _Float16* __restrict__ vloc_f16)
{
  __shared__ alignas(16) _Float16 w1T[HID_N * DIN_N];   // [n][k]
  __shared__ alignas(16) _Float16 w2T[HID_N * HID_N];
  __shared__ alignas(16) _Float16 vpT[LAT_N * HID_N];
  __shared__ float ccs[NC_N * LAT_N];
  __shared__ float b1s[HID_N], b2s[HID_N], vbs[LAT_N];
  __shared__ alignas(16) _Float16 stage[2][16 * HID_N];
  __shared__ float vbuf[2][16 * LAT_N];

  const int tid = threadIdx.x;
  for (int i = tid; i < HID_N * DIN_N; i += 64) { int n = i / DIN_N, k = i % DIN_N; w1T[i] = (_Float16)fe_w1[k * HID_N + n]; }
  for (int i = tid; i < HID_N * HID_N; i += 64) { int n = i / HID_N, k = i % HID_N; w2T[i] = (_Float16)fe_w2[k * HID_N + n]; }
  for (int i = tid; i < LAT_N * HID_N; i += 64) { int n = i / HID_N, k = i % HID_N; vpT[i] = (_Float16)vp_w[k * LAT_N + n]; }
  for (int i = tid; i < NC_N * LAT_N; i += 64) ccs[i] = cc[i];
  for (int i = tid; i < HID_N; i += 64) { b1s[i] = fe_b1[i]; b2s[i] = fe_b2[i]; }
  for (int i = tid; i < LAT_N; i += 64) vbs[i] = vp_b[i];
  __syncthreads();

  const int wave = tid >> 5, lane = tid & 31;
  const int half = (lane < 16) ? 0 : 8;
  const int col0 = lane & 15;
  const int row_base = blockIdx.x * 64 + wave * 32;

  for (int sub = 0; sub < 2; ++sub) {
    const int row0 = row_base + sub * 16;

    // GEMM1: x(16x32) @ fe_w1 -> gelu -> stage
    v16h ax = load_A_f32(x + (size_t)row0 * DIN_N, DIN_N, lane);
    float hc[8][8];
#pragma unroll
    for (int t = 0; t < 8; ++t) {
      v8f c = {};
      v16h b = load_B_h16(w1T, DIN_N, t * 16, 0, lane);
      c = wmma_f16(ax, b, c);
      int col = t * 16 + col0;
#pragma unroll
      for (int j = 0; j < 8; ++j) hc[t][j] = gelu_f(c[j] + b1s[col]);
    }
#pragma unroll
    for (int t = 0; t < 8; ++t)
#pragma unroll
      for (int j = 0; j < 8; ++j)
        stage[wave][(j + half) * HID_N + t * 16 + col0] = (_Float16)hc[t][j];
    __syncthreads();

    // GEMM2: h1(16x128) @ fe_w2 -> gelu
#pragma unroll
    for (int t = 0; t < 8; ++t) {
      v8f c = {};
#pragma unroll
      for (int kk = 0; kk < HID_N; kk += 32) {
        v16h aa = load_A_h16(&stage[wave][kk], HID_N, lane);
        v16h b  = load_B_h16(w2T, HID_N, t * 16, kk, lane);
        c = wmma_f16(aa, b, c);
      }
      int col = t * 16 + col0;
#pragma unroll
      for (int j = 0; j < 8; ++j) hc[t][j] = gelu_f(c[j] + b2s[col]);
    }
    __syncthreads();
#pragma unroll
    for (int t = 0; t < 8; ++t)
#pragma unroll
      for (int j = 0; j < 8; ++j)
        stage[wave][(j + half) * HID_N + t * 16 + col0] = (_Float16)hc[t][j];
    __syncthreads();

    // GEMM3: h2(16x128) @ vp_w -> v (16x32)
#pragma unroll
    for (int t = 0; t < 2; ++t) {
      v8f c = {};
#pragma unroll
      for (int kk = 0; kk < HID_N; kk += 32) {
        v16h aa = load_A_h16(&stage[wave][kk], HID_N, lane);
        v16h b  = load_B_h16(vpT, HID_N, t * 16, kk, lane);
        c = wmma_f16(aa, b, c);
      }
      int col = t * 16 + col0;
#pragma unroll
      for (int j = 0; j < 8; ++j) vbuf[wave][(j + half) * LAT_N + col] = c[j] + vbs[col];
    }
    __syncthreads();

    // Per-row epilogue: scores, softmax, argmax, c_bar, v_local
    if (lane < 16) {
      const int r = lane;
      const size_t grow = (size_t)row0 + r;
      float vv[LAT_N];
#pragma unroll
      for (int l = 0; l < LAT_N; ++l) vv[l] = vbuf[wave][r * LAT_N + l];
      float sc[NC_N];
      float mx = -3.0e38f; int am = 0;
#pragma unroll
      for (int c = 0; c < NC_N; ++c) {
        float s = 0.f;
#pragma unroll
        for (int l = 0; l < LAT_N; ++l) s += vv[l] * ccs[c * LAT_N + l];
        s *= 0.17677669529663688f;  // 1/sqrt(32)
        sc[c] = s;
        if (s > mx) { mx = s; am = c; }
      }
      float se = 0.f;
#pragma unroll
      for (int c = 0; c < NC_N; ++c) { sc[c] = expf(sc[c] - mx); se += sc[c]; }
      float inv = 1.0f / se;
#pragma unroll
      for (int c = 0; c < NC_N; ++c) { sc[c] *= inv; enc_out[grow * NC_N + c] = sc[c]; }
      kch_out[grow] = am;
#pragma unroll
      for (int l = 0; l < LAT_N; ++l) {
        float cb = 0.f;
#pragma unroll
        for (int c = 0; c < NC_N; ++c) cb += sc[c] * ccs[c * LAT_N + l];
        cbar_out[grow * LAT_N + l] = cb;
        float vl = vv[l] - cb;
        vloc_f32[grow * LAT_N + l] = vl;
        vloc_f16[grow * LAT_N + l] = (_Float16)vl;
      }
    }
    __syncthreads();
  }
}

// ---------------------------------------------------------------------------
// K2: VQ — distance GEMM + argmin + z_q blend + delta + loss partials
//     256 threads = 8 waves x 16 rows = 128 rows/block; f16 chart staged via
//     async global->LDS copies (ASYNCcnt) when available.
// ---------------------------------------------------------------------------
__global__ __launch_bounds__(256) void k2_vq(
    const _Float16* __restrict__ cb_h, const float* __restrict__ cbn_g,
    const float* __restrict__ enc_rw,
    const float* __restrict__ vloc_f32, const _Float16* __restrict__ vloc_f16,
    _Float16* __restrict__ delta_h, float* __restrict__ zqb_out,
    float* __restrict__ loss_partial)
{
  __shared__ alignas(16) _Float16 cb[KC_N * LAT_N];  // one chart, f16 (32 KB)
  __shared__ alignas(16) float cbn[KC_N];
  __shared__ float red[256];

  const int tid = threadIdx.x, wave = tid >> 5, lane = tid & 31;
  const int half = (lane < 16) ? 0 : 8;
  const int col0 = lane & 15;
  const int row0 = blockIdx.x * 128 + wave * 16;

  float vl0[8], vl1[8], vnorm[8];
#pragma unroll
  for (int j = 0; j < 8; ++j) {
    size_t r = (size_t)row0 + j + half;
    vl0[j] = vloc_f32[r * LAT_N + col0];
    vl1[j] = vloc_f32[r * LAT_N + col0 + 16];
    vnorm[j] = vl0[j] * vl0[j] + vl1[j] * vl1[j];
  }
#pragma unroll
  for (int m = 1; m < 16; m <<= 1)
#pragma unroll
    for (int j = 0; j < 8; ++j) vnorm[j] += __shfl_xor(vnorm[j], m, 32);

  v16h a = load_A_h16(vloc_f16 + (size_t)row0 * LAT_N, LAT_N, lane);

  float zqb0[8], zqb1[8];
#pragma unroll
  for (int j = 0; j < 8; ++j) { zqb0[j] = 0.f; zqb1[j] = 0.f; }
  float lossacc = 0.f;

  for (int c = 0; c < NC_N; ++c) {
    __syncthreads();  // all waves done reading previous chart
#ifdef HAVE_ASYNC_LDS
    {
      const char* gsrc = (const char*)(cb_h + (size_t)c * KC_N * LAT_N);
      char* ldst = (char*)cb;
      for (int i = tid * 16; i < KC_N * LAT_N * 2; i += 256 * 16)
        __builtin_amdgcn_global_load_async_to_lds_b128(
            (i4v*)(gsrc + i), (i4v*)(ldst + i), 0, 0);
      const char* nsrc = (const char*)(cbn_g + (size_t)c * KC_N);
      if (tid < (KC_N * 4) / 16)
        __builtin_amdgcn_global_load_async_to_lds_b128(
            (i4v*)(nsrc + tid * 16), (i4v*)((char*)cbn + tid * 16), 0, 0);
      __builtin_amdgcn_s_wait_asynccnt(0);
    }
#else
    {
      const uint4* gsrc = (const uint4*)(cb_h + (size_t)c * KC_N * LAT_N);
      uint4* ldst = (uint4*)cb;
      for (int i = tid; i < (KC_N * LAT_N * 2) / 16; i += 256) ldst[i] = gsrc[i];
      const float* nsrc = cbn_g + (size_t)c * KC_N;
      for (int i = tid; i < KC_N; i += 256) cbn[i] = nsrc[i];
    }
#endif
    __syncthreads();

    float minv[8]; int mini[8];
#pragma unroll
    for (int j = 0; j < 8; ++j) { minv[j] = 3.4e38f; mini[j] = 0; }

    for (int t = 0; t < KC_N / 16; ++t) {
      v16h b = load_B_h16(cb, LAT_N, t * 16, 0, lane);
      v8f d = {};
      d = wmma_f16(a, b, d);
      int code = t * 16 + col0;
      float nn = cbn[code];
#pragma unroll
      for (int j = 0; j < 8; ++j) {
        float val = nn - 2.0f * d[j];
        if (val < minv[j]) { minv[j] = val; mini[j] = code; }
      }
    }
    // min+argmin reduce across the 16 lanes of each half (first-occurrence)
#pragma unroll
    for (int m = 1; m < 16; m <<= 1) {
#pragma unroll
      for (int j = 0; j < 8; ++j) {
        float ov = __shfl_xor(minv[j], m, 32);
        int   oi = __shfl_xor(mini[j], m, 32);
        if (ov < minv[j] || (ov == minv[j] && oi < mini[j])) { minv[j] = ov; mini[j] = oi; }
      }
    }
#pragma unroll
    for (int j = 0; j < 8; ++j) {
      size_t r = (size_t)row0 + j + half;
      float enc = enc_rw[r * NC_N + c];
      float z0 = (float)cb[mini[j] * LAT_N + col0];
      float z1 = (float)cb[mini[j] * LAT_N + col0 + 16];
      zqb0[j] += enc * z0;
      zqb1[j] += enc * z1;
      delta_h[(r * NC_N + c) * LAT_N + col0]      = (_Float16)(vl0[j] - z0);
      delta_h[(r * NC_N + c) * LAT_N + col0 + 16] = (_Float16)(vl1[j] - z1);
      if (col0 == 0) lossacc += enc * (minv[j] + vnorm[j]);
    }
  }
#pragma unroll
  for (int j = 0; j < 8; ++j) {
    size_t r = (size_t)row0 + j + half;
    zqb_out[r * LAT_N + col0]      = zqb0[j];
    zqb_out[r * LAT_N + col0 + 16] = zqb1[j];
  }
  red[tid] = lossacc;
  __syncthreads();
  if (tid == 0) {
    float s = 0.f;
    for (int i = 0; i < 256; ++i) s += red[i];
    loss_partial[blockIdx.x] = s;
  }
}

// ---------------------------------------------------------------------------
// K3: soft-field MLP per (b,c) + z_n blend + z_geo/z_tex/tanh + dec_rw softmax
// ---------------------------------------------------------------------------
__global__ __launch_bounds__(64) void k3_softfield(
    const float* __restrict__ sf_w1, const float* __restrict__ sf_b1,
    const float* __restrict__ sf_w2, const float* __restrict__ sf_b2,
    const float* __restrict__ lr_w,  const float* __restrict__ lr_b,
    const float* __restrict__ enc_rw, const _Float16* __restrict__ delta_h,
    const float* __restrict__ vloc_f32, const float* __restrict__ zqb,
    const float* __restrict__ cbar,
    float* __restrict__ zn_out, float* __restrict__ zgeo_out,
    float* __restrict__ decrw_out,
    _Float16* __restrict__ zg_h, _Float16* __restrict__ zt_h)
{
  __shared__ alignas(16) _Float16 w1T[MID_N * LAT_N];      // [n=16][k=32]
  __shared__ alignas(16) _Float16 w2T[LAT_N * (2 * MID_N)];// [n=32][k=32], K>=16 zero
  __shared__ float b1s[MID_N], b2s[LAT_N];
  __shared__ float lrs[LAT_N * NC_N], lrb[NC_N];
  __shared__ alignas(16) _Float16 tstage[2][16 * 32];
  __shared__ float encs[2][16 * NC_N];
  __shared__ float zgbuf[2][16 * LAT_N];

  const int tid = threadIdx.x, wave = tid >> 5, lane = tid & 31;
  const int half = (lane < 16) ? 0 : 8;
  const int col0 = lane & 15;
  const int row0 = blockIdx.x * 32 + wave * 16;

  for (int i = tid; i < MID_N * LAT_N; i += 64) { int n = i / LAT_N, k = i % LAT_N; w1T[i] = (_Float16)sf_w1[k * MID_N + n]; }
  for (int i = tid; i < LAT_N * 2 * MID_N; i += 64) {
    int n = i / (2 * MID_N), k = i % (2 * MID_N);
    w2T[i] = (k < MID_N) ? (_Float16)sf_w2[k * LAT_N + n] : (_Float16)0.f;
  }
  for (int i = tid; i < MID_N; i += 64) b1s[i] = sf_b1[i];
  for (int i = tid; i < LAT_N; i += 64) b2s[i] = sf_b2[i];
  for (int i = tid; i < LAT_N * NC_N; i += 64) lrs[i] = lr_w[i];
  for (int i = tid; i < NC_N; i += 64) lrb[i] = lr_b[i];
  // zero the padded K half of tstage (never rewritten)
  for (int i = tid; i < 2 * 16 * 16; i += 64) {
    int w = i / 256, rem = i % 256;
    tstage[w][(rem / 16) * 32 + 16 + (rem % 16)] = (_Float16)0.f;
  }
  // stage enc_rw tiles
  for (int i = lane; i < 16 * NC_N; i += 32)
    encs[wave][i] = enc_rw[((size_t)row0 + i / NC_N) * NC_N + (i % NC_N)];
  __syncthreads();

  float zacc[2][8];
#pragma unroll
  for (int t = 0; t < 2; ++t)
#pragma unroll
    for (int j = 0; j < 8; ++j) zacc[t][j] = 0.f;

  for (int c = 0; c < NC_N; ++c) {
    v16h a = load_A_h16(delta_h + ((size_t)row0 * NC_N + c) * LAT_N, NC_N * LAT_N, lane);
    v16h b1 = load_B_h16(w1T, LAT_N, 0, 0, lane);
    v8f t1 = {};
    t1 = wmma_f16(a, b1, t1);
#pragma unroll
    for (int j = 0; j < 8; ++j)
      tstage[wave][(j + half) * 32 + col0] = (_Float16)gelu_f(t1[j] + b1s[col0]);
    __syncthreads();
    v16h a2 = load_A_h16(&tstage[wave][0], 32, lane);
#pragma unroll
    for (int t2 = 0; t2 < 2; ++t2) {
      v16h b2 = load_B_h16(w2T, 2 * MID_N, t2 * 16, 0, lane);
      v8f z = {};
      z = wmma_f16(a2, b2, z);
      int col = t2 * 16 + col0;
#pragma unroll
      for (int j = 0; j < 8; ++j)
        zacc[t2][j] += encs[wave][(j + half) * NC_N + c] * (z[j] + b2s[col]);
    }
    __syncthreads();
  }

#pragma unroll
  for (int t = 0; t < 2; ++t) {
    int col = t * 16 + col0;
#pragma unroll
    for (int j = 0; j < 8; ++j) {
      size_t r = (size_t)row0 + j + half;
      size_t idx = r * LAT_N + col;
      float zn = zacc[t][j];
      float qb = zqb[idx];
      float cb = cbar[idx];
      float vl = vloc_f32[idx];
      float zgeo = cb + qb + zn;
      float zg = tanhf(zgeo);
      float zt = tanhf(vl - qb - zn);
      zn_out[idx] = zn;
      zgeo_out[idx] = zgeo;
      zg_h[idx] = (_Float16)zg;
      zt_h[idx] = (_Float16)zt;
      zgbuf[wave][(j + half) * LAT_N + col] = zg;
    }
  }
  __syncthreads();
  if (lane < 16) {
    const int r = lane;
    const size_t grow = (size_t)row0 + r;
    float sc[NC_N];
    float mx = -3.0e38f;
#pragma unroll
    for (int c = 0; c < NC_N; ++c) {
      float s = lrb[c];
#pragma unroll
      for (int l = 0; l < LAT_N; ++l) s += zgbuf[wave][r * LAT_N + l] * lrs[l * NC_N + c];
      sc[c] = s;
      if (s > mx) mx = s;
    }
    float se = 0.f;
#pragma unroll
    for (int c = 0; c < NC_N; ++c) { sc[c] = expf(sc[c] - mx); se += sc[c]; }
    float inv = 1.0f / se;
#pragma unroll
    for (int c = 0; c < NC_N; ++c) decrw_out[grow * NC_N + c] = sc[c] * inv;
  }
}

// ---------------------------------------------------------------------------
// K4: decoder bank (dec_rw-weighted) + renderer + skip + texture -> x_hat
//     2 waves x 2 subtiles x 16 rows = 64 rows/block
// ---------------------------------------------------------------------------
__global__ __launch_bounds__(64) void k4_decoder(
    const float* __restrict__ dec_w, const float* __restrict__ dec_b,
    const float* __restrict__ decrw,
    const float* __restrict__ ren_w1, const float* __restrict__ ren_b1,
    const float* __restrict__ ren_w2, const float* __restrict__ ren_b2,
    const float* __restrict__ skip_w, const float* __restrict__ skip_b,
    const float* __restrict__ tex_w,  const float* __restrict__ tex_b,
    const float* __restrict__ tex_scale,
    const _Float16* __restrict__ zg_h, const _Float16* __restrict__ zt_h,
    float* __restrict__ xhat)
{
  __shared__ alignas(16) _Float16 r1T[HID_N * HID_N];  // 32 KB
  __shared__ float dbs[NC_N * HID_N];                  // 4 KB
  __shared__ float rb1[HID_N], rb2[LAT_N], skb[LAT_N], txb[LAT_N];
  __shared__ alignas(16) _Float16 buf0[2][16 * HID_N];
  __shared__ alignas(16) _Float16 buf1[2][16 * HID_N];

  const int tid = threadIdx.x, wave = tid >> 5, lane = tid & 31;
  const int half = (lane < 16) ? 0 : 8;
  const int col0 = lane & 15;
  const int row_base = blockIdx.x * 64 + wave * 32;

  for (int i = tid; i < HID_N * HID_N; i += 64) { int n = i / HID_N, k = i % HID_N; r1T[i] = (_Float16)ren_w1[k * HID_N + n]; }
  for (int i = tid; i < NC_N * HID_N; i += 64) dbs[i] = dec_b[i];
  for (int i = tid; i < HID_N; i += 64) rb1[i] = ren_b1[i];
  for (int i = tid; i < LAT_N; i += 64) { rb2[i] = ren_b2[i]; skb[i] = skip_b[i]; txb[i] = tex_b[i]; }
  __syncthreads();

  const float ts = tex_scale[0];

  for (int sub = 0; sub < 2; ++sub) {
    const int row0 = row_base + sub * 16;
    v16h az = load_A_h16(zg_h + (size_t)row0 * LAT_N, LAT_N, lane);

    // h_global = sum_c dec_rw * (zg @ dec_w[c]^T + dec_b[c])
    float hacc[8][8];
#pragma unroll
    for (int t = 0; t < 8; ++t)
#pragma unroll
      for (int j = 0; j < 8; ++j) hacc[t][j] = 0.f;

    for (int c = 0; c < NC_N; ++c) {
      float drw[8];
#pragma unroll
      for (int j = 0; j < 8; ++j) drw[j] = decrw[((size_t)row0 + j + half) * NC_N + c];
#pragma unroll
      for (int t = 0; t < 8; ++t) {
        v16h b = load_B_f32c(dec_w + (size_t)c * HID_N * LAT_N, LAT_N, t * 16, 0, lane);
        v8f d = {};
        d = wmma_f16(az, b, d);
        float bb = dbs[c * HID_N + t * 16 + col0];
#pragma unroll
        for (int j = 0; j < 8; ++j) hacc[t][j] += drw[j] * (d[j] + bb);
      }
    }
#pragma unroll
    for (int t = 0; t < 8; ++t)
#pragma unroll
      for (int j = 0; j < 8; ++j) {
        int idx = (j + half) * HID_N + t * 16 + col0;
        buf0[wave][idx] = (_Float16)hacc[t][j];           // h (for skip)
        buf1[wave][idx] = (_Float16)gelu_f(hacc[t][j]);   // gelu(h) (for ren1)
      }
    __syncthreads();

    // ren1: gelu(h) @ ren_w1 -> gelu
    float racc[8][8];
#pragma unroll
    for (int t = 0; t < 8; ++t) {
      v8f c = {};
#pragma unroll
      for (int kk = 0; kk < HID_N; kk += 32) {
        v16h aa = load_A_h16(&buf1[wave][kk], HID_N, lane);
        v16h b  = load_B_h16(r1T, HID_N, t * 16, kk, lane);
        c = wmma_f16(aa, b, c);
      }
#pragma unroll
      for (int j = 0; j < 8; ++j) racc[t][j] = gelu_f(c[j] + rb1[t * 16 + col0]);
    }
    __syncthreads();
#pragma unroll
    for (int t = 0; t < 8; ++t)
#pragma unroll
      for (int j = 0; j < 8; ++j)
        buf1[wave][(j + half) * HID_N + t * 16 + col0] = (_Float16)racc[t][j];
    __syncthreads();

    // out = ren2(buf1) + skip(buf0) + ts * tex(zt) + biases
    v16h at = load_A_h16(zt_h + (size_t)row0 * LAT_N, LAT_N, lane);
#pragma unroll
    for (int t = 0; t < 2; ++t) {
      v8f cacc = {};
#pragma unroll
      for (int kk = 0; kk < HID_N; kk += 32) {
        v16h aa = load_A_h16(&buf1[wave][kk], HID_N, lane);
        v16h b  = load_B_f32s(ren_w2, LAT_N, t * 16, kk, lane);
        cacc = wmma_f16(aa, b, cacc);
        v16h a0 = load_A_h16(&buf0[wave][kk], HID_N, lane);
        v16h bs = load_B_f32s(skip_w, LAT_N, t * 16, kk, lane);
        cacc = wmma_f16(a0, bs, cacc);
      }
      v16h bt = load_B_f32s(tex_w, LAT_N, t * 16, 0, lane);
      v8f tacc = {};
      tacc = wmma_f16(at, bt, tacc);
      int col = t * 16 + col0;
#pragma unroll
      for (int j = 0; j < 8; ++j) {
        size_t r = (size_t)row0 + j + half;
        xhat[r * LAT_N + col] = cacc[j] + rb2[col] + skb[col] + ts * (tacc[j] + txb[col]);
      }
    }
    __syncthreads();  // protect buf0/buf1 before next subtile overwrites
  }
}

// ---------------------------------------------------------------------------
// K5: deterministic loss reduction
// ---------------------------------------------------------------------------
__global__ __launch_bounds__(256) void k5_loss(const float* __restrict__ partials,
                                               int n, float* __restrict__ out)
{
  __shared__ float red[256];
  float s = 0.f;
  for (int i = threadIdx.x; i < n; i += 256) s += partials[i];
  red[threadIdx.x] = s;
  __syncthreads();
  for (int st = 128; st > 0; st >>= 1) {
    if (threadIdx.x < st) red[threadIdx.x] += red[threadIdx.x + st];
    __syncthreads();
  }
  if (threadIdx.x == 0)
    out[0] = 1.25f / (float)((size_t)B_N * LAT_N) * red[0];
}

// ---------------------------------------------------------------------------
extern "C" void kernel_launch(void* const* d_in, const int* in_sizes, int n_in,
                              void* d_out, int out_size, void* d_ws, size_t ws_size,
                              hipStream_t stream) {
  const float* x        = (const float*)d_in[0];
  const float* fe_w1    = (const float*)d_in[1];
  const float* fe_b1    = (const float*)d_in[2];
  const float* fe_w2    = (const float*)d_in[3];
  const float* fe_b2    = (const float*)d_in[4];
  const float* vp_w     = (const float*)d_in[5];
  const float* vp_b     = (const float*)d_in[6];
  const float* cc       = (const float*)d_in[7];
  const float* codebook = (const float*)d_in[8];
  const float* sf_w1    = (const float*)d_in[9];
  const float* sf_b1    = (const float*)d_in[10];
  const float* sf_w2    = (const float*)d_in[11];
  const float* sf_b2    = (const float*)d_in[12];
  const float* dec_w    = (const float*)d_in[13];
  const float* dec_b    = (const float*)d_in[14];
  const float* lr_w     = (const float*)d_in[15];
  const float* lr_b     = (const float*)d_in[16];
  const float* tex_w    = (const float*)d_in[17];
  const float* tex_b    = (const float*)d_in[18];
  const float* tex_s    = (const float*)d_in[19];
  const float* ren_w1   = (const float*)d_in[20];
  const float* ren_b1   = (const float*)d_in[21];
  const float* ren_w2   = (const float*)d_in[22];
  const float* ren_b2   = (const float*)d_in[23];
  const float* skip_w   = (const float*)d_in[24];
  const float* skip_b   = (const float*)d_in[25];

  // output layout: x_hat, vq_loss, enc_rw, dec_rw, K_chart, z_geo, z_n, c_bar
  float* out   = (float*)d_out;
  float* xhat  = out;
  float* vq    = out + 2097152;
  float* enc   = out + 2097153;
  float* drw   = out + 2621441;
  int*   kch   = (int*)(out + 3145729);
  float* zgeo  = out + 3211265;
  float* zn    = out + 5308417;
  float* cbar  = out + 7405569;

  char* ws = (char*)d_ws;
  float*    vloc_f32 = (float*)(ws + 0);
  _Float16* vloc_f16 = (_Float16*)(ws + 8388608ull);
  _Float16* delta_h  = (_Float16*)(ws + 12582912ull);
  float*    zqb      = (float*)(ws + 46137344ull);
  _Float16* zg_h     = (_Float16*)(ws + 54525952ull);
  _Float16* zt_h     = (_Float16*)(ws + 58720256ull);
  float*    partials = (float*)(ws + 62914560ull);   // 512 floats
  _Float16* cb_h     = (_Float16*)(ws + 62922752ull); // 256 KB
  float*    cbn_g    = (float*)(ws + 62988288ull);    // 16 KB

  k0_prep<<<(NC_N * KC_N + 255) / 256, 256, 0, stream>>>(codebook, cb_h, cbn_g);
  k1_encoder<<<B_N / 64, 64, 0, stream>>>(x, fe_w1, fe_b1, fe_w2, fe_b2, vp_w,
                                          vp_b, cc, enc, kch, cbar,
                                          vloc_f32, vloc_f16);
  k2_vq<<<B_N / 128, 256, 0, stream>>>(cb_h, cbn_g, enc, vloc_f32, vloc_f16,
                                       delta_h, zqb, partials);
  k3_softfield<<<B_N / 32, 64, 0, stream>>>(sf_w1, sf_b1, sf_w2, sf_b2, lr_w,
                                            lr_b, enc, delta_h, vloc_f32, zqb,
                                            cbar, zn, zgeo, drw, zg_h, zt_h);
  k4_decoder<<<B_N / 64, 64, 0, stream>>>(dec_w, dec_b, drw, ren_w1, ren_b1,
                                          ren_w2, ren_b2, skip_w, skip_b, tex_w,
                                          tex_b, tex_s, zg_h, zt_h, xhat);
  k5_loss<<<1, 256, 0, stream>>>(partials, B_N / 128, vq);
}
